// PromptBlock_39934605918362
// MI455X (gfx1250) — matrix operand order repaired and checked by reading, not compiled, not run on hardware
//
#include <hip/hip_runtime.h>
#include <hip/hip_bf16.h>
#include <cstdint>

// ---------------- problem constants ----------------
// B=16, N=1024, D=768, H=12, HD=64, P=16, FF=3072, KV=P+N=1040, M=B*N=16384

typedef __attribute__((ext_vector_type(16))) __bf16 v16bf;
typedef __attribute__((ext_vector_type(8)))  float  v8f;

union FragU { v16bf v; uint4 u[2]; };

__device__ __forceinline__ v16bf frag_ld(const __bf16* p, int off1) {
  FragU f;
  f.u[0] = *(const uint4*)p;
  f.u[1] = *(const uint4*)(p + off1);
  return f.v;
}

__device__ __forceinline__ v16bf frag_ldk(const __bf16* p, int ld) {
  v16bf v;
#pragma unroll
  for (int s = 0; s < 16; ++s) v[s] = p[((s & 7) + ((s >> 3) << 4)) * ld];
  return v;
}

__device__ __forceinline__ v8f wmma_bf16(v16bf a, v16bf b, v8f c) {
  return __builtin_amdgcn_wmma_f32_16x16x32_bf16(false, a, false, b, (short)0, c,
                                                 false, false);
}

__device__ __forceinline__ __bf16 f2bf(float f) { return (__bf16)f; }

// ---------------- workspace layout (bytes) ----------------
static constexpr size_t SZ_H    = (size_t)16384 * 768 * 2;       // 25,165,824
static constexpr size_t SZ_Q    = (size_t)192 * 1024 * 64 * 2;   // 25,165,824
static constexpr size_t SZ_KT   = (size_t)192 * 64 * 1040 * 2;   // 25,559,040
static constexpr size_t SZ_V    = (size_t)192 * 1040 * 64 * 2;   // 25,559,040
static constexpr size_t SZ_O    = SZ_H;

static constexpr size_t OFF_H    = 0;
static constexpr size_t OFF_Q    = OFF_H + SZ_H;
static constexpr size_t OFF_KT   = OFF_Q + SZ_Q;
static constexpr size_t OFF_V    = OFF_KT + SZ_KT;
static constexpr size_t OFF_O    = OFF_V + SZ_V;
static constexpr size_t OFF_M    = OFF_Q;            // m overlaps dead q/kT/v/o
static constexpr size_t OFF_WQKV = OFF_O + SZ_O;
static constexpr size_t OFF_WOUT = OFF_WQKV + (size_t)768 * 2304 * 2;
static constexpr size_t OFF_WFC1 = OFF_WOUT + (size_t)768 * 768 * 2;
static constexpr size_t OFF_WFC2 = OFF_WFC1 + (size_t)768 * 3072 * 2;

// ---------------- weight transpose + fp32->bf16 ----------------
// w[N,K] fp32 row-major  ->  wt[K,N] bf16 row-major
__global__ __launch_bounds__(256)
void wtrans_kernel(const float* __restrict__ w, __bf16* __restrict__ wt,
                   int N, int K) {
  __shared__ float tile[32][33];
  const int k0 = blockIdx.x * 32;
  const int n0 = blockIdx.y * 32;
  const int tx = threadIdx.x & 31;
  const int ty = threadIdx.x >> 5;  // 0..7
#pragma unroll
  for (int i = 0; i < 4; ++i)
    tile[ty + 8 * i][tx] = w[(size_t)(n0 + ty + 8 * i) * K + k0 + tx];
  __syncthreads();
#pragma unroll
  for (int i = 0; i < 4; ++i)
    wt[(size_t)(k0 + ty + 8 * i) * N + n0 + tx] = f2bf(tile[tx][ty + 8 * i]);
}

// ---------------- prompt scatter ----------------
// prompt [B,2,P,H,64] fp32 -> kT[bh,64,1040] (cols 0..15), v[bh,1040,64] (rows 0..15)
__global__ __launch_bounds__(256)
void prompt_kernel(const float* __restrict__ pr, __bf16* __restrict__ kt,
                   __bf16* __restrict__ v) {
  const int idx = blockIdx.x * 256 + threadIdx.x;  // < 393216 exactly
  const int d = idx & 63;
  const int h = (idx >> 6) % 12;
  const int p = (idx / 768) & 15;
  const int t = (idx / 12288) & 1;
  const int b = idx / 24576;
  const float val = pr[idx];
  const int bh = b * 12 + h;
  if (t == 0) kt[((size_t)bh * 64 + d) * 1040 + p] = f2bf(val);
  else        v[((size_t)bh * 1040 + p) * 64 + d]  = f2bf(val);
}

// ---------------- layernorm -> bf16 ----------------
__global__ __launch_bounds__(256)
void ln_kernel(const float* __restrict__ x, const float* __restrict__ gm,
               const float* __restrict__ bt, __bf16* __restrict__ out) {
  const int lane = threadIdx.x & 31;
  const int row = blockIdx.x * 8 + (threadIdx.x >> 5);
  const float* xr = x + (size_t)row * 768;
  float v[24];
  float s = 0.0f;
#pragma unroll
  for (int i = 0; i < 24; ++i) { v[i] = xr[i * 32 + lane]; s += v[i]; }
#pragma unroll
  for (int d = 16; d >= 1; d >>= 1) s += __shfl_xor(s, d, 32);
  const float mu = s * (1.0f / 768.0f);
  float q = 0.0f;
#pragma unroll
  for (int i = 0; i < 24; ++i) { float t = v[i] - mu; q += t * t; }
#pragma unroll
  for (int d = 16; d >= 1; d >>= 1) q += __shfl_xor(q, d, 32);
  const float rstd = rsqrtf(q * (1.0f / 768.0f) + 1e-5f);
  __bf16* orow = out + (size_t)row * 768;
#pragma unroll
  for (int i = 0; i < 24; ++i) {
    const int col = i * 32 + lane;
    orow[col] = f2bf((v[i] - mu) * rstd * gm[col] + bt[col]);
  }
}

// ---------------- tiled WMMA GEMM: C[M,N] = A[M,K] * Bt[K,N] + bias ----------------
enum { EPI_QKV = 0, EPI_ADD = 1, EPI_GELU = 2 };

template <int EPI, int KD, int ND>
__global__ __launch_bounds__(256)
void gemm_kernel(const __bf16* __restrict__ A, const __bf16* __restrict__ Bt,
                 const float* __restrict__ bias, const float* resid, float* outf,
                 __bf16* outq, __bf16* outkt, __bf16* outv, __bf16* outbf) {
  constexpr int LDA = 40;   // 80B rows: 16B aligned, bank-spread
  constexpr int LDB = 136;  // 272B rows
  constexpr int NK = KD / 32;
  __shared__ __align__(16) __bf16 sA[2][128 * LDA];
  __shared__ __align__(16) __bf16 sB[2][32 * LDB];

  const int tid = threadIdx.x;
  const int lane = tid & 31;
  const int wid = tid >> 5;
  const int m0 = blockIdx.y * 128;
  const int n0 = blockIdx.x * 128;
  const int wm0 = (wid >> 1) * 32;
  const int wn0 = (wid & 1) * 64;

  uint4 ra[2], rb[2];
  auto gload = [&](int kt) {
#pragma unroll
    for (int t = 0; t < 2; ++t) {
      const int i = tid + t * 256;
      ra[t] = *(const uint4*)(A + (size_t)(m0 + (i >> 2)) * KD + kt * 32 + (i & 3) * 8);
      rb[t] = *(const uint4*)(Bt + (size_t)(kt * 32 + (i >> 4)) * ND + n0 + (i & 15) * 8);
    }
  };
  auto sstore = [&](int buf) {
#pragma unroll
    for (int t = 0; t < 2; ++t) {
      const int i = tid + t * 256;
      *(uint4*)&sA[buf][(i >> 2) * LDA + (i & 3) * 8] = ra[t];
      *(uint4*)&sB[buf][(i >> 4) * LDB + (i & 15) * 8] = rb[t];
    }
  };

  v8f acc[2][4] = {};

  gload(0);
  sstore(0);
  for (int kt = 0; kt < NK; ++kt) {
    __syncthreads();
    const int cur = kt & 1;
    if (kt + 1 < NK) gload(kt + 1);
    const __bf16* pa = &sA[cur][(wm0 + (lane & 15)) * LDA + ((lane >> 4) << 3)];
    const v16bf af0 = frag_ld(pa, 16);
    const v16bf af1 = frag_ld(pa + 16 * LDA, 16);
    const __bf16* pb = &sB[cur][((lane >> 4) << 3) * LDB + wn0 + (lane & 15)];
#pragma unroll
    for (int ni = 0; ni < 4; ++ni) {
      const v16bf bf = frag_ldk(pb + ni * 16, LDB);
      acc[0][ni] = wmma_bf16(af0, bf, acc[0][ni]);
      acc[1][ni] = wmma_bf16(af1, bf, acc[1][ni]);
    }
    if (kt + 1 < NK) sstore((kt + 1) & 1);
  }

  // epilogue: C layout -> VGPR r, lane l: row = r + 8*(l>>4), col = l&15
  const int g = lane >> 4, c = lane & 15;
#pragma unroll
  for (int mi = 0; mi < 2; ++mi) {
#pragma unroll
    for (int ni = 0; ni < 4; ++ni) {
      const int col = n0 + wn0 + ni * 16 + c;
      const float bcol = bias[col];
#pragma unroll
      for (int r = 0; r < 8; ++r) {
        const int row = m0 + wm0 + mi * 16 + r + 8 * g;
        float val = acc[mi][ni][r] + bcol;
        if constexpr (EPI == EPI_QKV) {
          const int which = col / 768;
          const int rem = col - which * 768;
          const int head = rem >> 6, d = rem & 63;
          const int b = row >> 10, n = row & 1023;
          const int bh = b * 12 + head;
          if (which == 0)
            outq[((size_t)bh * 1024 + n) * 64 + d] = f2bf(val * 0.125f);  // pre-scale q
          else if (which == 1)
            outkt[((size_t)bh * 64 + d) * 1040 + 16 + n] = f2bf(val);     // transposed k
          else
            outv[((size_t)bh * 1040 + 16 + n) * 64 + d] = f2bf(val);
        } else if constexpr (EPI == EPI_ADD) {
          const size_t idx = (size_t)row * ND + col;
          outf[idx] = resid[idx] + val;
        } else {  // QuickGELU
          const float gz = val / (1.0f + __expf(-1.702f * val));
          outbf[(size_t)row * ND + col] = f2bf(gz);
        }
      }
    }
  }
}

// ---------------- flash attention ----------------
// Q [bh,1024,64] (pre-scaled), KT [bh,64,1040], V [bh,1040,64] -> O [B,1024,768]
// 8 full kv tiles of 128 (no masking) + peeled 16-wide tail tile.
__global__ __launch_bounds__(128)
void attn_kernel(const __bf16* __restrict__ Q, const __bf16* __restrict__ KTt,
                 const __bf16* __restrict__ V, __bf16* __restrict__ O) {
  constexpr int LQ = 72, LK = 136, LV = 72, LP = 136;
  __shared__ __align__(16) __bf16 sQ[64 * LQ];
  __shared__ __align__(16) __bf16 sKT[64 * LK];
  __shared__ __align__(16) __bf16 sV[128 * LV];
  __shared__ __align__(16) __bf16 sP[64 * LP];

  const int tid = threadIdx.x;
  const int lane = tid & 31;
  const int wid = tid >> 5;  // 0..3, each owns 16 q-rows
  const int bh = blockIdx.y;
  const int q0 = blockIdx.x * 64;

  const __bf16* Qg = Q + (size_t)bh * 1024 * 64 + (size_t)q0 * 64;
  const __bf16* Kg = KTt + (size_t)bh * 64 * 1040;
  const __bf16* Vg = V + (size_t)bh * 1040 * 64;

  // Q tile: 64 rows x 64 halves = 512 uint4
#pragma unroll
  for (int t = 0; t < 4; ++t) {
    const int i = tid + t * 128;
    *(uint4*)&sQ[(i >> 3) * LQ + (i & 7) * 8] =
        *(const uint4*)(Qg + (size_t)(i >> 3) * 64 + (i & 7) * 8);
  }

  const int g = lane >> 4, c = lane & 15;
  float mi[8], li[8];
  v8f oacc[4] = {};
#pragma unroll
  for (int r = 0; r < 8; ++r) { mi[r] = -3.0e38f; li[r] = 0.0f; }

  const __bf16* pa = &sQ[(wid * 16 + (lane & 15)) * LQ + ((lane >> 4) << 3)];
  const __bf16* pp = &sP[(wid * 16 + (lane & 15)) * LP + ((lane >> 4) << 3)];

  // ---------- 8 full kv tiles (kv 0..1023), no masking ----------
  for (int j = 0; j < 8; ++j) {
    const int kvb = j * 128;
    __syncthreads();  // previous tile fully consumed
#pragma unroll
    for (int t = 0; t < 8; ++t) {
      const int i = tid + t * 128;
      *(uint4*)&sKT[(i >> 4) * LK + (i & 15) * 8] =
          *(const uint4*)(Kg + (size_t)(i >> 4) * 1040 + kvb + (i & 15) * 8);
    }
#pragma unroll
    for (int t = 0; t < 8; ++t) {
      const int i = tid + t * 128;
      *(uint4*)&sV[(i >> 3) * LV + (i & 7) * 8] =
          *(const uint4*)(Vg + (size_t)(kvb + (i >> 3)) * 64 + (i & 7) * 8);
    }
    __syncthreads();

    // S = Q * K^T  (16 x 128 per wave)
    v8f sacc[8] = {};
#pragma unroll
    for (int ks = 0; ks < 2; ++ks) {
      const v16bf aq = frag_ld(pa + ks * 32, 16);
      const __bf16* pb = &sKT[(ks * 32 + ((lane >> 4) << 3)) * LK + (lane & 15)];
#pragma unroll
      for (int t = 0; t < 8; ++t) {
        const v16bf bk = frag_ldk(pb + t * 16, LK);
        sacc[t] = wmma_bf16(aq, bk, sacc[t]);
      }
    }

    // per-row running max
    float mt[8];
#pragma unroll
    for (int r = 0; r < 8; ++r) mt[r] = sacc[0][r];
#pragma unroll
    for (int t = 1; t < 8; ++t)
#pragma unroll
      for (int r = 0; r < 8; ++r) mt[r] = fmaxf(mt[r], sacc[t][r]);
#pragma unroll
    for (int d = 8; d >= 1; d >>= 1)
#pragma unroll
      for (int r = 0; r < 8; ++r) mt[r] = fmaxf(mt[r], __shfl_xor(mt[r], d, 16));

    float alpha[8];
#pragma unroll
    for (int r = 0; r < 8; ++r) {
      const float mn = fmaxf(mi[r], mt[r]);
      alpha[r] = __expf(mi[r] - mn);
      mi[r] = mn;
    }

    // P = exp(S - m); row sums; stash P (bf16) in wave-private LDS slice
    float rs[8];
#pragma unroll
    for (int r = 0; r < 8; ++r) rs[r] = 0.0f;
#pragma unroll
    for (int t = 0; t < 8; ++t) {
#pragma unroll
      for (int r = 0; r < 8; ++r) {
        const float p = __expf(sacc[t][r] - mi[r]);
        rs[r] += p;
        sP[(wid * 16 + r + 8 * g) * LP + t * 16 + c] = f2bf(p);
      }
    }
#pragma unroll
    for (int d = 8; d >= 1; d >>= 1)
#pragma unroll
      for (int r = 0; r < 8; ++r) rs[r] += __shfl_xor(rs[r], d, 16);
#pragma unroll
    for (int r = 0; r < 8; ++r) li[r] = li[r] * alpha[r] + rs[r];
#pragma unroll
    for (int t = 0; t < 4; ++t)
#pragma unroll
      for (int r = 0; r < 8; ++r) oacc[t][r] *= alpha[r];

    // O += P * V   (V[kv][64] gathered per column, k strided by LV)
#pragma unroll
    for (int ks = 0; ks < 4; ++ks) {
      const v16bf ap = frag_ld(pp + ks * 32, 16);
      const __bf16* pv = &sV[(ks * 32 + ((lane >> 4) << 3)) * LV + (lane & 15)];
#pragma unroll
      for (int t = 0; t < 4; ++t) {
        const v16bf bv = frag_ldk(pv + t * 16, LV);
        oacc[t] = wmma_bf16(ap, bv, oacc[t]);
      }
    }
  }

  // ---------- peeled tail tile: kv 1024..1039 (exactly 16 columns) ----------
  {
    const int kvb = 1024;
    __syncthreads();
    {  // KT tail: 64 rows x 16 halves = 128 uint4
      const int r = tid >> 1, ch = (tid & 1) * 8;
      *(uint4*)&sKT[r * LK + ch] = *(const uint4*)(Kg + (size_t)r * 1040 + kvb + ch);
    }
    {  // V tail: 16 rows x 64 halves = 128 uint4
      const int r = tid >> 3, ch = (tid & 7) * 8;
      *(uint4*)&sV[r * LV + ch] = *(const uint4*)(Vg + (size_t)(kvb + r) * 64 + ch);
    }
    __syncthreads();

    // S tail: single 16x16 sub-tile, all columns valid
    v8f sacc0 = {};
#pragma unroll
    for (int ks = 0; ks < 2; ++ks) {
      const v16bf aq = frag_ld(pa + ks * 32, 16);
      const v16bf bk = frag_ldk(&sKT[(ks * 32 + ((lane >> 4) << 3)) * LK + (lane & 15)], LK);
      sacc0 = wmma_bf16(aq, bk, sacc0);
    }

    float mt[8];
#pragma unroll
    for (int r = 0; r < 8; ++r) mt[r] = sacc0[r];
#pragma unroll
    for (int d = 8; d >= 1; d >>= 1)
#pragma unroll
      for (int r = 0; r < 8; ++r) mt[r] = fmaxf(mt[r], __shfl_xor(mt[r], d, 16));

    float alpha[8];
#pragma unroll
    for (int r = 0; r < 8; ++r) {
      const float mn = fmaxf(mi[r], mt[r]);
      alpha[r] = __expf(mi[r] - mn);
      mi[r] = mn;
    }

    float rs[8];
#pragma unroll
    for (int r = 0; r < 8; ++r) {
      const float p = __expf(sacc0[r] - mi[r]);
      rs[r] = p;
      const int prow = (wid * 16 + r + 8 * g) * LP;
      sP[prow + c] = f2bf(p);
      sP[prow + 16 + c] = f2bf(0.0f);  // zero pad: P k-step covers kv 0..31
    }
#pragma unroll
    for (int d = 8; d >= 1; d >>= 1)
#pragma unroll
      for (int r = 0; r < 8; ++r) rs[r] += __shfl_xor(rs[r], d, 16);
#pragma unroll
    for (int r = 0; r < 8; ++r) li[r] = li[r] * alpha[r] + rs[r];
#pragma unroll
    for (int t = 0; t < 4; ++t)
#pragma unroll
      for (int r = 0; r < 8; ++r) oacc[t][r] *= alpha[r];

    // O += P * V : single k-step (kv 0..31; 16..31 multiply exact zeros)
    const v16bf ap = frag_ld(pp, 16);
#pragma unroll
    for (int t = 0; t < 4; ++t) {
      const v16bf bv = frag_ldk(&sV[((lane >> 4) << 3) * LV + (lane & 15) + t * 16], LV);
      oacc[t] = wmma_bf16(ap, bv, oacc[t]);
    }
  }

  const int b = bh / 12, h = bh - (bh / 12) * 12;
#pragma unroll
  for (int r = 0; r < 8; ++r) {
    const float inv = 1.0f / li[r];
    const int n = q0 + wid * 16 + r + 8 * g;
#pragma unroll
    for (int t = 0; t < 4; ++t)
      O[((size_t)b * 1024 + n) * 768 + h * 64 + t * 16 + c] = f2bf(oacc[t][r] * inv);
  }
}

// ---------------- launcher ----------------
extern "C" void kernel_launch(void* const* d_in, const int* in_sizes, int n_in,
                              void* d_out, int out_size, void* d_ws, size_t ws_size,
                              hipStream_t stream) {
  (void)in_sizes; (void)n_in; (void)out_size; (void)ws_size;
  const float* x      = (const float*)d_in[0];
  const float* prompt = (const float*)d_in[1];
  const float* qkv_w  = (const float*)d_in[2];
  const float* qkv_b  = (const float*)d_in[3];
  const float* out_w  = (const float*)d_in[4];
  const float* out_b  = (const float*)d_in[5];
  const float* ln1_g  = (const float*)d_in[6];
  const float* ln1_b  = (const float*)d_in[7];
  const float* ln2_g  = (const float*)d_in[8];
  const float* ln2_b  = (const float*)d_in[9];
  const float* fc1_w  = (const float*)d_in[10];
  const float* fc1_b  = (const float*)d_in[11];
  const float* fc2_w  = (const float*)d_in[12];
  const float* fc2_b  = (const float*)d_in[13];
  float* outp = (float*)d_out;
  char* ws = (char*)d_ws;

  __bf16* hbuf  = (__bf16*)(ws + OFF_H);
  __bf16* qbuf  = (__bf16*)(ws + OFF_Q);
  __bf16* ktbuf = (__bf16*)(ws + OFF_KT);
  __bf16* vbuf  = (__bf16*)(ws + OFF_V);
  __bf16* obuf  = (__bf16*)(ws + OFF_O);
  __bf16* mbuf  = (__bf16*)(ws + OFF_M);
  __bf16* wqkvT = (__bf16*)(ws + OFF_WQKV);
  __bf16* woutT = (__bf16*)(ws + OFF_WOUT);
  __bf16* wfc1T = (__bf16*)(ws + OFF_WFC1);
  __bf16* wfc2T = (__bf16*)(ws + OFF_WFC2);

  // weights -> bf16 transposed [K,N]
  wtrans_kernel<<<dim3(24, 72), 256, 0, stream>>>(qkv_w, wqkvT, 2304, 768);
  wtrans_kernel<<<dim3(24, 24), 256, 0, stream>>>(out_w, woutT, 768, 768);
  wtrans_kernel<<<dim3(24, 96), 256, 0, stream>>>(fc1_w, wfc1T, 3072, 768);
  wtrans_kernel<<<dim3(96, 24), 256, 0, stream>>>(fc2_w, wfc2T, 768, 3072);

  // prompt KV into kT / v prefixes
  prompt_kernel<<<1536, 256, 0, stream>>>(prompt, ktbuf, vbuf);

  // h = LN1(x)
  ln_kernel<<<2048, 256, 0, stream>>>(x, ln1_g, ln1_b, hbuf);

  // qkv = h @ qkv_w^T + b  (scatter into q / kT / v)
  gemm_kernel<EPI_QKV, 768, 2304><<<dim3(18, 128), 256, 0, stream>>>(
      hbuf, wqkvT, qkv_b, nullptr, nullptr, qbuf, ktbuf, vbuf, nullptr);

  // o = softmax(q kT) v
  attn_kernel<<<dim3(16, 192), 128, 0, stream>>>(qbuf, ktbuf, vbuf, obuf);

  // x1 = x + o @ out_w^T + b
  gemm_kernel<EPI_ADD, 768, 768><<<dim3(6, 128), 256, 0, stream>>>(
      obuf, woutT, out_b, x, outp, nullptr, nullptr, nullptr, nullptr);

  // h = LN2(x1)
  ln_kernel<<<2048, 256, 0, stream>>>(outp, ln2_g, ln2_b, hbuf);

  // m = quickgelu(h @ fc1_w^T + b)
  gemm_kernel<EPI_GELU, 768, 3072><<<dim3(24, 128), 256, 0, stream>>>(
      hbuf, wfc1T, fc1_b, nullptr, nullptr, nullptr, nullptr, nullptr, mbuf);

  // out = x1 + m @ fc2_w^T + b
  gemm_kernel<EPI_ADD, 3072, 768><<<dim3(6, 128), 256, 0, stream>>>(
      mbuf, wfc2T, fc2_b, outp, outp, nullptr, nullptr, nullptr, nullptr);
}
